// GConvGRURegression_29437705847598
// MI455X (gfx1250) — matrix operand order repaired and checked by reading, not compiled, not run on hardware
//
#include <hip/hip_runtime.h>
#include <hip/hip_bf16.h>

typedef __attribute__((ext_vector_type(16))) __bf16 v16bf;
typedef __attribute__((ext_vector_type(8)))  __bf16 v8bf;
typedef __attribute__((ext_vector_type(8)))  float  v8f;

#define B_HALVES (2 * 6 * 4 * 32 * 16)   // 24576 bf16 = 49152 bytes of packed weights

// ---------------- degree: deg[src] += ew  ----------------
__global__ void k_deg(const int* __restrict__ src, const float* __restrict__ ew,
                      float* __restrict__ deg, int E) {
  int e = blockIdx.x * blockDim.x + threadIdx.x;
  if (e < E) atomicAdd(&deg[src[e]], ew[e]);
}

// ---------------- edge norm: -rsqrt(deg[s])*ew*rsqrt(deg[d]) ----------------
__global__ void k_norm(const int* __restrict__ src, const int* __restrict__ dst,
                       const float* __restrict__ ew, const float* __restrict__ deg,
                       float* __restrict__ norm, int E) {
  int e = blockIdx.x * blockDim.x + threadIdx.x;
  if (e >= E) return;
  float ds = deg[src[e]], dd = deg[dst[e]];
  float is = ds > 0.f ? rsqrtf(ds) : 0.f;
  float id = dd > 0.f ? rsqrtf(dd) : 0.f;
  norm[e] = -is * ew[e] * id;
}

// ---------------- SpMM: out[dst] += norm * in[src]  (16 threads/edge, float4) ----
__global__ void k_prop(const int* __restrict__ src, const int* __restrict__ dst,
                       const float* __restrict__ norm, const float* __restrict__ tin,
                       float* __restrict__ tout, int E) {
  long t = (long)blockIdx.x * blockDim.x + threadIdx.x;
  if (t >= (long)E * 16) return;
  int e = (int)(t >> 4);
  int f = ((int)t & 15) * 4;
  int s = src[e], d = dst[e];
  float w = norm[e];
  const float4 v = *reinterpret_cast<const float4*>(tin + (long)s * 64 + f);
  float* o = tout + (long)d * 64 + f;
  atomicAdd(o + 0, w * v.x);
  atomicAdd(o + 1, w * v.y);
  atomicAdd(o + 2, w * v.z);
  atomicAdd(o + 3, w * v.w);
}

// ---------------- build bf16 A = [Tx0 | Tx1 | 2*P2 - Tx0]  (N x 192) ----------------
__global__ void k_combine(const float* __restrict__ x, const float* __restrict__ t1,
                          const float* __restrict__ p2, __bf16* __restrict__ A, int N) {
  long t = (long)blockIdx.x * blockDim.x + threadIdx.x;
  if (t >= (long)N * 64) return;
  int n = (int)(t >> 6), f = (int)t & 63;
  float x0 = x[t];
  __bf16* row = A + (long)n * 192;
  row[f]        = (__bf16)x0;
  row[64 + f]   = (__bf16)t1[t];
  row[128 + f]  = (__bf16)(2.f * p2[t] - x0);
}

// ---------------- pack W stacks (192x64 each, 2 gates) into WMMA B-lane layout ------
// B[( ((g*6+kb)*4+c)*32 + lane )*16 + j] = W[g][kb*32 + koff + (j<8 ? j : 8+j)][c*16 + lane%16]
__global__ void k_packB(const float* __restrict__ Wz, const float* __restrict__ Wh,
                        __bf16* __restrict__ B) {
  int tid = blockIdx.x * blockDim.x + threadIdx.x;
  if (tid >= B_HALVES) return;
  int idx = tid;
  int j    = idx & 15; idx >>= 4;
  int lane = idx & 31; idx >>= 5;
  int c    = idx & 3;  idx >>= 2;
  int kb   = idx % 6;  idx /= 6;
  int g    = idx;
  int koff = (lane < 16) ? 0 : 8;
  int K = kb * 32 + koff + (j < 8 ? j : 8 + j);
  int n = c * 16 + (lane & 15);
  const float* W = g ? Wh : Wz;
  B[tid] = (__bf16)W[K * 64 + n];
}

// ---------------- fused: two 16x192x64 GEMMs + GRU blend + 64->1 linear + sigmoid ---
// 8 waves/block; packed B staged once per block into LDS via gfx1250 async copy.
__global__ void __launch_bounds__(256)
k_gemm_fused(const __bf16* __restrict__ A, const __bf16* __restrict__ Bg,
             const float* __restrict__ b_xz, const float* __restrict__ b_hz,
             const float* __restrict__ b_xh, const float* __restrict__ b_hh,
             const float* __restrict__ W_lin, const float* __restrict__ b_lin,
             float* __restrict__ out, int N) {
  __shared__ __align__(16) __bf16 Bl[B_HALVES];   // 49152 bytes

  const int tid = threadIdx.x;

  // --- stage packed B -> LDS with async copies (ASYNCcnt path) ---
  {
    unsigned ldsBase = (unsigned)(size_t)(void*)&Bl[0];
    unsigned long long gBase = (unsigned long long)(size_t)Bg;
    #pragma unroll
    for (int k = 0; k < 12; ++k) {                 // 256 thr * 12 * 16B = 49152 B
      unsigned ofs = (unsigned)(tid + k * 256) * 16u;
      unsigned l = ldsBase + ofs;
      unsigned long long g = gBase + ofs;
      asm volatile("global_load_async_to_lds_b128 %0, %1, off"
                   :: "v"(l), "v"(g) : "memory");
    }
    asm volatile("s_wait_asynccnt 0x0" ::: "memory");
  }
  __syncthreads();

  const int lane = tid & 31;
  const int wave = tid >> 5;
  const int ntiles = (N + 15) / 16;
  const int tile = blockIdx.x * 8 + wave;
  if (tile >= ntiles) return;

  const int base = tile * 16;
  const int m    = lane & 15;
  const int off  = (lane >> 4) * 8;               // A: lanes>=16 take K+8 halves
  int row = base + m; if (row >= N) row = N - 1;
  const __bf16* arow = A + (long)row * 192;

  v8f zero = {};
  v8f acc[2][4];
  #pragma unroll
  for (int g = 0; g < 2; ++g)
    #pragma unroll
    for (int c = 0; c < 4; ++c) acc[g][c] = zero;

  if (base + 16 + m < N)                          // speculative prefetch of next tile
    __builtin_prefetch(A + (long)(base + 16 + m) * 192, 0, 1);

  #pragma unroll
  for (int kb = 0; kb < 6; ++kb) {
    const __bf16* ap = arow + kb * 32;
    v8bf alo = *reinterpret_cast<const v8bf*>(ap + off);
    v8bf ahi = *reinterpret_cast<const v8bf*>(ap + 16 + off);
    v16bf a = __builtin_shufflevector(alo, ahi, 0,1,2,3,4,5,6,7,8,9,10,11,12,13,14,15);
    #pragma unroll
    for (int g = 0; g < 2; ++g) {
      #pragma unroll
      for (int c = 0; c < 4; ++c) {
        const __bf16* bp = &Bl[(size_t)(((g * 6 + kb) * 4 + c) * 32 + lane) * 16];
        v8bf blo = *reinterpret_cast<const v8bf*>(bp);        // ds_load_b128
        v8bf bhi = *reinterpret_cast<const v8bf*>(bp + 8);
        v16bf bb = __builtin_shufflevector(blo, bhi, 0,1,2,3,4,5,6,7,8,9,10,11,12,13,14,15);
        acc[g][c] = __builtin_amdgcn_wmma_f32_16x16x32_bf16(
            false, a, false, bb, (short)0, acc[g][c], false, false);
      }
    }
  }

  // epilogue: C layout -> vgpr r holds (M = r + 8*(lane>=16), N = c*16 + lane%16)
  const int nlo = lane & 15;
  float bz[4], bh[4], wl[4];
  #pragma unroll
  for (int c = 0; c < 4; ++c) {
    int n = c * 16 + nlo;
    bz[c] = b_xz[n] + b_hz[n];
    bh[c] = b_xh[n] + b_hh[n];
    wl[c] = W_lin[n];
  }
  const float bl = b_lin[0];

  #pragma unroll
  for (int r = 0; r < 8; ++r) {
    float partial = 0.f;
    #pragma unroll
    for (int c = 0; c < 4; ++c) {
      float z  = 1.f / (1.f + __expf(-(acc[0][c][r] + bz[c])));
      float ht = tanhf(acc[1][c][r] + bh[c]);
      partial += (1.f - z) * ht * wl[c];          // H = (1-Z)*H~ since H0 == 0
    }
    #pragma unroll
    for (int msk = 1; msk < 16; msk <<= 1)        // reduce within each 16-lane half
      partial += __shfl_xor(partial, msk, 32);
    if (nlo == 0) {
      int node = base + r + ((lane >> 4) << 3);
      if (node < N) out[node] = 1.f / (1.f + __expf(-(partial + bl)));
    }
  }
}

extern "C" void kernel_launch(void* const* d_in, const int* in_sizes, int n_in,
                              void* d_out, int out_size, void* d_ws, size_t ws_size,
                              hipStream_t stream) {
  (void)n_in; (void)out_size; (void)ws_size;
  const float* x     = (const float*)d_in[0];
  const int*   ei    = (const int*)d_in[1];
  const float* ew    = (const float*)d_in[2];
  const float* W_xz  = (const float*)d_in[3];
  const float* b_xz  = (const float*)d_in[4];
  const float* b_hz  = (const float*)d_in[6];
  const float* W_xh  = (const float*)d_in[11];
  const float* b_xh  = (const float*)d_in[12];
  const float* b_hh  = (const float*)d_in[14];
  const float* W_lin = (const float*)d_in[15];
  const float* b_lin = (const float*)d_in[16];

  const int N = in_sizes[0] / 64;
  const int E = in_sizes[1] / 2;
  const int* src = ei;
  const int* dst = ei + E;

  char* ws = (char*)d_ws;
  size_t o = 0;
  auto alloc = [&](size_t bytes) {
    char* p = ws + o; o += (bytes + 255) & ~(size_t)255; return p;
  };
  float*  deg  = (float*)alloc((size_t)N * 4);
  float*  norm = (float*)alloc((size_t)E * 4);
  float*  Tx1  = (float*)alloc((size_t)N * 64 * 4);
  float*  P2   = (float*)alloc((size_t)N * 64 * 4);
  __bf16* Abf  = (__bf16*)alloc((size_t)N * 192 * 2);
  __bf16* Bpk  = (__bf16*)alloc((size_t)B_HALVES * 2);

  hipMemsetAsync(deg, 0, (size_t)N * 4, stream);
  hipMemsetAsync(Tx1, 0, (size_t)N * 64 * 4, stream);
  hipMemsetAsync(P2,  0, (size_t)N * 64 * 4, stream);

  const int tb = 256;
  k_deg <<<(E + tb - 1) / tb, tb, 0, stream>>>(src, ew, deg, E);
  k_norm<<<(E + tb - 1) / tb, tb, 0, stream>>>(src, dst, ew, deg, norm, E);

  long pt = (long)E * 16;
  k_prop<<<(int)((pt + tb - 1) / tb), tb, 0, stream>>>(src, dst, norm, x,   Tx1, E);
  k_prop<<<(int)((pt + tb - 1) / tb), tb, 0, stream>>>(src, dst, norm, Tx1, P2,  E);

  long ct = (long)N * 64;
  k_combine<<<(int)((ct + tb - 1) / tb), tb, 0, stream>>>(x, Tx1, P2, Abf, N);
  k_packB  <<<(B_HALVES + tb - 1) / tb, tb, 0, stream>>>(W_xz, W_xh, Bpk);

  int ntiles = (N + 15) / 16;
  int nblocks = (ntiles + 7) / 8;
  k_gemm_fused<<<nblocks, 256, 0, stream>>>(Abf, Bpk, b_xz, b_hz, b_xh, b_hh,
                                            W_lin, b_lin, (float*)d_out, N);
}
